// NODETC_47347719471584
// MI455X (gfx1250) — matrix operand order repaired and checked by reading, not compiled, not run on hardware
//
#include <hip/hip_runtime.h>

typedef _Float16 half_t;
typedef __attribute__((ext_vector_type(16))) _Float16 v16h;
typedef __attribute__((ext_vector_type(8)))  _Float16 v8h;
typedef __attribute__((ext_vector_type(8)))  float    v8f;

#define B_N    4096
#define T_N    64
#define OBS_N  16
#define STAT_N 8
#define LAT_N  32
#define HID_N  64
#define K_N    8

#define WAVES_PER_WG 4
#define THREADS      (WAVES_PER_WG * 32)
#define ROWS_PER_WG  (WAVES_PER_WG * 16)        // 64 batch rows / WG
#define WG_PER_K     (B_N / ROWS_PER_WG)        // 64
#define ACT_STRIDE   72   // halves: 144B rows -> 16B aligned b128, bank-conflict-spread

// ---- register-resident weight fragments (B-matrix layout) -----------------
struct Wreg {
    v16h W1[4];        // layer1: K=32,  N-tiles 0..3
    v16h W2[4][2];     // layer2: K=64 (2 chunks), N-tiles 0..3
    v16h W3[2][2];     // layer3: K=64, N-tiles 0..1
};

// per-lane bias scalars (splat into C at use)
struct Breg {
    float b1[4];
    float b2[4];
    float b3[2];
};

__device__ __forceinline__ v8f wmma16(v16h a, v16h b, v8f c) {
    // v_wmma_f32_16x16x32_f16 : D = A(16x32 f16) * B(32x16 f16) + C(f32)
    return __builtin_amdgcn_wmma_f32_16x16x32_f16(false, a, false, b,
                                                  (short)0, c, false, false);
}

__device__ __forceinline__ v8f splat8(float v) {
    v8f c;
#pragma unroll
    for (int i = 0; i < 8; ++i) c[i] = v;
    return c;
}

__device__ __forceinline__ v8f fmadd8(v8f a, float s, v8f b) {  // a + s*b
    v8f r;
#pragma unroll
    for (int i = 0; i < 8; ++i) r[i] = fmaf(s, b[i], a[i]);
    return r;
}

// Branch-free tanh producing f16 for LDS staging (prefers HW v_tanh).
__device__ __forceinline__ half_t tanh_h(float v) {
#if __has_builtin(__builtin_amdgcn_tanh_f16)
    return __builtin_amdgcn_tanh_f16((half_t)v);
#elif __has_builtin(__builtin_amdgcn_tanhf)
    return (half_t)__builtin_amdgcn_tanhf(v);
#else
    // branch-free rational fallback on native exp
    float ax = __builtin_fabsf(v);
    float e  = __expf(-2.0f * ax);
    float r  = 1.0f - 2.0f * e / (1.0f + e);
    return (half_t)__builtin_copysignf(r, v);
#endif
}

// A-fragment (16x32 f16) from row-major LDS staging buffer [16][ACT_STRIDE].
__device__ __forceinline__ v16h loadAfrag(const half_t* act, int lane, int kc) {
    const int m    = lane & 15;
    const int col0 = kc * 32 + ((lane >> 4) & 1) * 8;
    const half_t* p = act + m * ACT_STRIDE + col0;
    v8h lo = *(const v8h*)(p);        // ds_load_b128
    v8h hi = *(const v8h*)(p + 16);   // ds_load_b128
    v16h r;
#pragma unroll
    for (int i = 0; i < 8; ++i) { r[i] = lo[i]; r[i + 8] = hi[i]; }
    return r;
}

// B-fragment (32x16 f16) from LDS-resident *transposed* weights Wt[N][Kdim].
__device__ __forceinline__ v16h loadBfrag(const half_t* wt, int Kdim, int nt,
                                          int kc, int lane) {
    const int n  = nt * 16 + (lane & 15);
    const int k0 = kc * 32 + ((lane >> 4) & 1) * 8;
    const half_t* p = wt + n * Kdim + k0;
    v8h lo = *(const v8h*)(p);
    v8h hi = *(const v8h*)(p + 16);
    v16h r;
#pragma unroll
    for (int i = 0; i < 8; ++i) { r[i] = lo[i]; r[i + 8] = hi[i]; }
    return r;
}

// Store one 16x16 f32 C/D tile through tanh to staging as f16.
__device__ __forceinline__ void storeTileTanh(half_t* act, int lane, int nt, v8f c) {
    const int n  = nt * 16 + (lane & 15);
    const int mb = ((lane >> 4) & 1) * 8;
#pragma unroll
    for (int r = 0; r < 8; ++r)
        act[(mb + r) * ACT_STRIDE + n] = tanh_h(c[r]);    // ds_store_b16
}

// Store one 16x16 f32 tile as f16 (no activation).
__device__ __forceinline__ void storeTileRaw(half_t* act, int lane, int nt, v8f c) {
    const int n  = nt * 16 + (lane & 15);
    const int mb = ((lane >> 4) & 1) * 8;
#pragma unroll
    for (int r = 0; r < 8; ++r)
        act[(mb + r) * ACT_STRIDE + n] = (half_t)c[r];
}

__device__ __forceinline__ void stageZ(half_t* act, int lane, v8f z0, v8f z1) {
    storeTileRaw(act, lane, 0, z0);
    storeTileRaw(act, lane, 1, z1);
}

// Global f32 [kValid x Ndim] row-major -> LDS f16 transposed [Ndim][Kdim]
__device__ __forceinline__ void loadWT(const float* g, half_t* s, int Kdim,
                                       int Ndim, int kValid, int tid) {
    const int total = Kdim * Ndim;
    for (int i = tid; i < total; i += THREADS) {
        int k = i / Ndim, n = i - k * Ndim;
        float v = (k < kValid) ? g[(size_t)k * Ndim + n] : 0.0f;
        s[n * Kdim + k] = (half_t)v;
    }
}

__device__ __forceinline__ void loadBiasLds(const float* g, float* s, int n, int tid) {
    for (int i = tid; i < n; i += THREADS) s[i] = g[i];
}

__device__ __forceinline__ void loadWreg(Wreg& w, const half_t* sW1t,
                                         const half_t* sW2t, const half_t* sW3t,
                                         int lane) {
#pragma unroll
    for (int nt = 0; nt < 4; ++nt) w.W1[nt] = loadBfrag(sW1t, LAT_N, nt, 0, lane);
#pragma unroll
    for (int nt = 0; nt < 4; ++nt)
#pragma unroll
        for (int kc = 0; kc < 2; ++kc)
            w.W2[nt][kc] = loadBfrag(sW2t, HID_N, nt, kc, lane);
#pragma unroll
    for (int nt = 0; nt < 2; ++nt)
#pragma unroll
        for (int kc = 0; kc < 2; ++kc)
            w.W3[nt][kc] = loadBfrag(sW3t, HID_N, nt, kc, lane);
}

__device__ __forceinline__ void loadBreg(Breg& br, const float* sB1,
                                         const float* sB2, const float* sB3,
                                         int lane) {
    const int nlo = lane & 15;
#pragma unroll
    for (int nt = 0; nt < 4; ++nt) br.b1[nt] = sB1[nt * 16 + nlo];
#pragma unroll
    for (int nt = 0; nt < 4; ++nt) br.b2[nt] = sB2[nt * 16 + nlo];
#pragma unroll
    for (int nt = 0; nt < 2; ++nt) br.b3[nt] = sB3[nt * 16 + nlo];
}

// 3-layer MLP; layer-1 A fragment supplied by caller (input already staged).
// Independent WMMAs issue together, then the branch-free tanh/stores follow.
__device__ __forceinline__ void mlpCoreA(const Wreg& w, const Breg& br,
                                         half_t* act, int lane, v16h a,
                                         v8f& o0, v8f& o1) {
    {   // layer 1: 32 -> 64, tanh
        v8f c[4];
#pragma unroll
        for (int nt = 0; nt < 4; ++nt)
            c[nt] = wmma16(a, w.W1[nt], splat8(br.b1[nt]));
#pragma unroll
        for (int nt = 0; nt < 4; ++nt)
            storeTileTanh(act, lane, nt, c[nt]);
    }
    {   // layer 2: 64 -> 64, tanh (in-place: read frags before stores)
        v16h a0 = loadAfrag(act, lane, 0);
        v16h a1 = loadAfrag(act, lane, 1);
        v8f c[4];
#pragma unroll
        for (int nt = 0; nt < 4; ++nt) {
            v8f t = wmma16(a0, w.W2[nt][0], splat8(br.b2[nt]));
            c[nt] = wmma16(a1, w.W2[nt][1], t);
        }
#pragma unroll
        for (int nt = 0; nt < 4; ++nt)
            storeTileTanh(act, lane, nt, c[nt]);
    }
    {   // layer 3: 64 -> 32, linear, stays in registers
        v16h a0 = loadAfrag(act, lane, 0);
        v16h a1 = loadAfrag(act, lane, 1);
        v8f c0 = wmma16(a0, w.W3[0][0], splat8(br.b3[0]));
        c0     = wmma16(a1, w.W3[0][1], c0);
        v8f c1 = wmma16(a0, w.W3[1][0], splat8(br.b3[1]));
        c1     = wmma16(a1, w.W3[1][1], c1);
        o0 = c0; o1 = c1;
    }
}

__device__ __forceinline__ void mlpEval(const Wreg& w, const Breg& br,
                                        half_t* act, int lane, v8f i0, v8f i1,
                                        v8f& o0, v8f& o1) {
    stageZ(act, lane, i0, i1);
    v16h a = loadAfrag(act, lane, 0);
    mlpCoreA(w, br, act, lane, a, o0, o1);
}

// preds[t] = z @ dec_W + dec_b from an already-loaded z A-fragment.
__device__ __forceinline__ void decodeA(float* outW, v16h a, v16h bdec, float bd,
                                        int lane, int tstep) {
    v8f c = wmma16(a, bdec, splat8(bd));
    const int n  = lane & 15;
    const int mb = ((lane >> 4) & 1) * 8;
#pragma unroll
    for (int r = 0; r < 8; ++r)
        outW[((size_t)(mb + r) * T_N + tstep) * OBS_N + n] = c[r];
}

__global__ __launch_bounds__(THREADS)
void node_rk4_wmma_kernel(const float* __restrict__ tg,
                          const float* __restrict__ x,
                          const float* __restrict__ zs,
                          const float* __restrict__ eW1, const float* __restrict__ eb1,
                          const float* __restrict__ eW2, const float* __restrict__ eb2,
                          const float* __restrict__ eW3, const float* __restrict__ eb3,
                          const float* __restrict__ oW1, const float* __restrict__ ob1,
                          const float* __restrict__ oW2, const float* __restrict__ ob2,
                          const float* __restrict__ oW3, const float* __restrict__ ob3,
                          const float* __restrict__ dW,  const float* __restrict__ db,
                          float* __restrict__ out) {
    __shared__ __align__(16) half_t sW1t[HID_N * LAT_N];   // Wt [64][32]
    __shared__ __align__(16) half_t sW2t[HID_N * HID_N];   // Wt [64][64]
    __shared__ __align__(16) half_t sW3t[LAT_N * HID_N];   // Wt [32][64]
    __shared__ __align__(16) half_t sDWt[OBS_N * LAT_N];   // Wt [16][32]
    __shared__ float sB1[HID_N], sB2[HID_N], sB3[LAT_N], sDB[OBS_N];
    __shared__ __align__(16) half_t sAct[WAVES_PER_WG][16 * ACT_STRIDE];

    const int tid  = threadIdx.x;
    const int lane = tid & 31;
    const int wave = tid >> 5;
    const int kIdx = blockIdx.x / WG_PER_K;
    const int bt   = blockIdx.x % WG_PER_K;
    const int b0   = bt * ROWS_PER_WG + wave * 16;
    half_t* act    = sAct[wave];

    // ---- phase 1: encoder (input padded 24->32 features) ----
    loadWT(eW1, sW1t, LAT_N, HID_N, OBS_N + STAT_N, tid);
    loadWT(eW2, sW2t, HID_N, HID_N, HID_N, tid);
    loadWT(eW3, sW3t, HID_N, LAT_N, HID_N, tid);
    loadWT(dW,  sDWt, LAT_N, OBS_N, LAT_N, tid);
    loadBiasLds(eb1, sB1, HID_N, tid);
    loadBiasLds(eb2, sB2, HID_N, tid);
    loadBiasLds(eb3, sB3, LAT_N, tid);
    loadBiasLds(db,  sDB, OBS_N, tid);
    __syncthreads();

    // stage [x[b,0,:], z[b,:], 0-pad] into this wave's A buffer (lanes 0..15 = rows)
    if (lane < 16) {
        const int b = b0 + lane;
        const float* xp = x + (size_t)b * T_N * OBS_N;
        const float* zp = zs + (size_t)b * STAT_N;
#pragma unroll
        for (int j = 0; j < OBS_N; ++j)
            act[lane * ACT_STRIDE + j] = (half_t)xp[j];
#pragma unroll
        for (int j = 0; j < STAT_N; ++j)
            act[lane * ACT_STRIDE + OBS_N + j] = (half_t)zp[j];
#pragma unroll
        for (int j = 0; j < STAT_N; ++j)
            act[lane * ACT_STRIDE + OBS_N + STAT_N + j] = (half_t)0.0f;
    }

    Wreg w;
    Breg br;
    loadWreg(w, sW1t, sW2t, sW3t, lane);
    loadBreg(br, sB1, sB2, sB3, lane);
    const v16h  bdec = loadBfrag(sDWt, LAT_N, 0, 0, lane);  // dec_W, register-resident
    const float bd   = sDB[lane & 15];

    v8f z0, z1;
    {
        v16h a = loadAfrag(act, lane, 0);
        mlpCoreA(w, br, act, lane, a, z0, z1);      // z0 = encoder output
    }

    __syncthreads();   // all waves done reading encoder weights from LDS

    // ---- phase 2: this cluster's ODE weights ----
    loadWT(oW1 + (size_t)kIdx * LAT_N * HID_N, sW1t, LAT_N, HID_N, LAT_N, tid);
    loadWT(oW2 + (size_t)kIdx * HID_N * HID_N, sW2t, HID_N, HID_N, HID_N, tid);
    loadWT(oW3 + (size_t)kIdx * HID_N * LAT_N, sW3t, HID_N, LAT_N, HID_N, tid);
    loadBiasLds(ob1 + kIdx * HID_N, sB1, HID_N, tid);
    loadBiasLds(ob2 + kIdx * HID_N, sB2, HID_N, tid);
    loadBiasLds(ob3 + kIdx * LAT_N, sB3, LAT_N, tid);
    __syncthreads();

    loadWreg(w, sW1t, sW2t, sW3t, lane);
    loadBreg(br, sB1, sB2, sB3, lane);

    float* outW = out + ((size_t)kIdx * B_N + b0) * T_N * OBS_N;

    // ---- fixed-step RK4; decoder shares the k1 stage's z A-fragment ----
    for (int s = 0; s < T_N - 1; ++s) {
        const float h = tg[s + 1] - tg[s];
        v8f ka, kb, ta, tb, acc0, acc1;

        stageZ(act, lane, z0, z1);
        {
            v16h a = loadAfrag(act, lane, 0);
            decodeA(outW, a, bdec, bd, lane, s);               // preds at t=s
            mlpCoreA(w, br, act, lane, a, ka, kb);             // k1 = f(z)
        }
        acc0 = fmadd8(z0, h * (1.0f / 6.0f), ka);
        acc1 = fmadd8(z1, h * (1.0f / 6.0f), kb);
        ta   = fmadd8(z0, 0.5f * h, ka);
        tb   = fmadd8(z1, 0.5f * h, kb);

        mlpEval(w, br, act, lane, ta, tb, ka, kb);             // k2
        acc0 = fmadd8(acc0, h * (1.0f / 3.0f), ka);
        acc1 = fmadd8(acc1, h * (1.0f / 3.0f), kb);
        ta   = fmadd8(z0, 0.5f * h, ka);
        tb   = fmadd8(z1, 0.5f * h, kb);

        mlpEval(w, br, act, lane, ta, tb, ka, kb);             // k3
        acc0 = fmadd8(acc0, h * (1.0f / 3.0f), ka);
        acc1 = fmadd8(acc1, h * (1.0f / 3.0f), kb);
        ta   = fmadd8(z0, h, ka);
        tb   = fmadd8(z1, h, kb);

        mlpEval(w, br, act, lane, ta, tb, ka, kb);             // k4
        z0 = fmadd8(acc0, h * (1.0f / 6.0f), ka);
        z1 = fmadd8(acc1, h * (1.0f / 6.0f), kb);
    }

    // final decode at t = T-1
    stageZ(act, lane, z0, z1);
    {
        v16h a = loadAfrag(act, lane, 0);
        decodeA(outW, a, bdec, bd, lane, T_N - 1);
    }
}

extern "C" void kernel_launch(void* const* d_in, const int* in_sizes, int n_in,
                              void* d_out, int out_size, void* d_ws, size_t ws_size,
                              hipStream_t stream) {
    (void)in_sizes; (void)n_in; (void)out_size; (void)d_ws; (void)ws_size;
    const float* tg  = (const float*)d_in[0];
    const float* x   = (const float*)d_in[1];
    const float* zs  = (const float*)d_in[2];
    const float* eW1 = (const float*)d_in[3];
    const float* eb1 = (const float*)d_in[4];
    const float* eW2 = (const float*)d_in[5];
    const float* eb2 = (const float*)d_in[6];
    const float* eW3 = (const float*)d_in[7];
    const float* eb3 = (const float*)d_in[8];
    const float* oW1 = (const float*)d_in[9];
    const float* ob1 = (const float*)d_in[10];
    const float* oW2 = (const float*)d_in[11];
    const float* ob2 = (const float*)d_in[12];
    const float* oW3 = (const float*)d_in[13];
    const float* ob3 = (const float*)d_in[14];
    const float* dW  = (const float*)d_in[15];
    const float* db  = (const float*)d_in[16];

    dim3 grid(K_N * WG_PER_K);   // 8 clusters * 64 batch tiles = 512 WGs
    dim3 block(THREADS);         // 4 waves * 16 batch rows each
    node_rk4_wmma_kernel<<<grid, block, 0, stream>>>(
        tg, x, zs, eW1, eb1, eW2, eb2, eW3, eb3,
        oW1, ob1, oW2, ob2, oW3, ob3, dW, db, (float*)d_out);
}
